// ReviewNet_53876069761211
// MI455X (gfx1250) — matrix operand order repaired and checked by reading, not compiled, not run on hardware
//
#include <hip/hip_runtime.h>

// ---------------- problem constants ----------------
#define BSZ   64
#define TENC  100
#define TDEC  50
#define LSZ   512
#define ESZ   512
#define HSZ   512
#define RSZ   8
#define VOUT  30000
#define L5    2560          // 5*L

typedef __attribute__((ext_vector_type(16))) _Float16 v16h;
typedef __attribute__((ext_vector_type(8)))  float    v8f;
typedef __attribute__((ext_vector_type(4)))  int      v4i;

#define AS1 __attribute__((address_space(1)))
#define AS3 __attribute__((address_space(3)))

#if defined(__AMDGCN__) && __has_builtin(__builtin_amdgcn_global_load_async_to_lds_b128) && __has_builtin(__builtin_amdgcn_s_wait_asynccnt)
#define HAVE_ASYNC_LDS 1
#else
#define HAVE_ASYNC_LDS 0
#endif

// ---------------- weight repack: f32 KxN -> f16 WMMA-fragment order, padded ----
// dst[((nt*KC + kc)*32 + lane)*16 + j] = (f16) src[(kc*32 + (lane>>4)*16 + j)*N + nt*16 + (lane&15)]
// Columns >= N are zero-filled so the GEMM K-loop needs no bounds checks.
__global__ void k_pack_w(const float* __restrict__ in, _Float16* __restrict__ out,
                         int N, int Npad, int K, int nslices) {
  size_t dstSlice = (size_t)Npad * K;
  size_t p = (size_t)blockIdx.x * blockDim.x + threadIdx.x;
  if (p >= dstSlice * (size_t)nslices) return;
  size_t sl = p / dstSlice;
  size_t q  = p - sl * dstSlice;
  int j    = (int)(q & 15);
  int lane = (int)((q >> 4) & 31);
  int KC   = K >> 5;
  int kc   = (int)((q >> 9) % (size_t)KC);
  int nt   = (int)(q / ((size_t)KC << 9));
  int k = (kc << 5) + ((lane >> 4) << 4) + j;
  int n = (nt << 4) + (lane & 15);
  float v = (n < N) ? in[sl * (size_t)N * K + (size_t)k * N + n] : 0.0f;
  out[p] = (_Float16)v;
}

// ---------------- embedding gather -> f16, out rows ordered [t*B + b] ----------
__global__ void k_embed(const int* __restrict__ tok, const float* __restrict__ emb,
                        _Float16* __restrict__ out, int T) {
  int row = blockIdx.x;                 // b*T + t
  int b = row / T, t = row % T;
  int col = blockIdx.y * blockDim.x + threadIdx.x;
  int tk = tok[row];
  out[((size_t)(t * BSZ + b)) * (size_t)ESZ + col] = (_Float16)emb[(size_t)tk * ESZ + col];
}

// ---------------- WMMA GEMM: C = A(f16) * Wp(f16 packed) + bias [+aux] ---------
// A: M x K f16 (lda). Wp: packed+padded (see k_pack_w). C: M x N f32 (ldc).
// Block = 128 threads (4 waves). Block tile: 16(M) x 256(N); wave tile: 16 x 64.
// A staged to LDS with GLOBAL_LOAD_ASYNC_TO_LDS_B128 (ASYNCcnt) when available.
// K-loop is branch-free (B zero-padded); N-bound applied only at stores.
// MODE: 0 = C = acc+bias ; 1 = += aux (addend) ; 2 = += C (accumulate)
template <int MODE>
__global__ void k_gemm_f16(const _Float16* __restrict__ A, int lda,
                           const _Float16* __restrict__ Wp, int N, int K,
                           const float* __restrict__ bias,
                           const float* __restrict__ aux,
                           float* __restrict__ C, int ldc) {
  __shared__ _Float16 lds_a[16 * 72];            // 16 x 64 chunk, stride 72 (16B rows)
  const int tid  = threadIdx.x;
  const int lane = tid & 31;
  const int wid  = __builtin_amdgcn_readfirstlane(tid >> 5);  // provably uniform
  const int lm   = lane & 15;
  const int kh   = lane >> 4;
  const int m0   = blockIdx.y << 4;
  const int ntbase = (blockIdx.x << 4) + (wid << 2); // first of this wave's 4 N-tiles
  const int KC = K >> 5;                         // # 32-wide chunks
  const int KO = K >> 6;                         // # 64-wide LDS stages

  // per-tile packed-B pointers (explicit global addrspace -> global_load_b128)
  const AS1 _Float16* bp[4];
#pragma unroll
  for (int u = 0; u < 4; ++u)
    bp[u] = (const AS1 _Float16*)(size_t)(Wp + ((size_t)(ntbase + u) * KC) * 512 +
                                          (size_t)lane * 16);

  // this thread's staging slot: one 16-byte (8-half) segment per stage
  const int am = tid >> 3;                       // row 0..15
  const int ak = (tid & 7) << 3;                 // col 0,8,..,56

  v8f acc[4] = {};

  for (int ko = 0; ko < KO; ++ko) {
#pragma unroll
    for (int u = 0; u < 4; ++u)
      __builtin_prefetch((const _Float16*)(size_t)bp[u] + 1024, 0, 1);
    __syncthreads();                             // previous stage fully consumed
    {
      const _Float16* src = A + (size_t)(m0 + am) * lda + (ko << 6) + ak;
      _Float16* dst = &lds_a[am * 72 + ak];
#if HAVE_ASYNC_LDS
      __builtin_amdgcn_global_load_async_to_lds_b128(
          (AS1 v4i*)(size_t)src, (AS3 v4i*)(unsigned)(size_t)dst, 0, 0);
      __builtin_amdgcn_s_wait_asynccnt(0);
#else
      *(uint4*)(void*)dst = *(const uint4*)(const void*)src;
#endif
    }
    __syncthreads();

#pragma unroll
    for (int half = 0; half < 2; ++half) {
      // A fragment (contiguous k = 16*kh + s, same order as the B packing) from LDS, b32 reads
      union { v16h v; unsigned u32[8]; } af;
#pragma unroll
      for (int i = 0; i < 8; ++i) {
        int klo = (half << 5) + (kh << 4) + (i << 1);
        af.u32[i] = *(const unsigned*)&lds_a[lm * 72 + klo];
      }
#pragma unroll
      for (int u = 0; u < 4; ++u) {
        v16h b = *(const AS1 v16h*)bp[u];
        acc[u] = __builtin_amdgcn_wmma_f32_16x16x32_f16(false, af.v, false, b,
                                                        (short)0, acc[u], false, false);
      }
#pragma unroll
      for (int u = 0; u < 4; ++u) bp[u] += 512;
    }
  }

  const int crow = m0 + (kh << 3);
#pragma unroll
  for (int u = 0; u < 4; ++u) {
    int nt = ntbase + u;
    if ((nt << 4) < N) {                          // scalar (wid is uniform)
      int ccol = (nt << 4) + lm;
      float bv = bias[ccol];
#pragma unroll
      for (int i = 0; i < 8; ++i) {
        size_t ci = (size_t)(crow + i) * ldc + ccol;
        float v = acc[u][i] + bv;
        if (MODE == 1) v += aux[ci];
        if (MODE == 2) v += C[ci];
        C[ci] = v;
      }
    }
  }
}

// ---------------- maxout-LSTM pointwise: s,c in f32; h out f16 -----------------
__device__ __forceinline__ float sigf(float x) { return 1.0f / (1.0f + __expf(-x)); }

__global__ void k_lstm(const float* __restrict__ s, float* __restrict__ c,
                       _Float16* __restrict__ h, _Float16* __restrict__ hcopy,
                       int copy_stride) {
  int idx = blockIdx.x * blockDim.x + threadIdx.x;   // b*L + l
  int b = idx >> 9, l = idx & 511;
  const float* sr = s + (size_t)b * L5;
  float ig = sigf(sr[l]);
  float fg = sigf(sr[LSZ + l]);
  float og = sigf(sr[2 * LSZ + l]);
  float tg = fmaxf(sr[3 * LSZ + l], sr[4 * LSZ + l]);
  float c2 = fg * c[idx] + ig * tg;
  float h2 = og * tanhf(c2);
  c[idx] = c2;
  _Float16 h16 = (_Float16)h2;
  h[idx] = h16;
  if (hcopy) hcopy[(size_t)b * copy_stride + l] = h16;
}

// ---------------- attention scores: e[b,a] = tanh(P1+q) . watt + batt ----------
__global__ void k_att_scores(const float* __restrict__ P1, const float* __restrict__ q,
                             const float* __restrict__ watt, const float* __restrict__ batt,
                             float* __restrict__ e, int A) {
  int row = blockIdx.x;          // b*A + a
  int b = row / A;
  int lane = threadIdx.x;
  float sum = 0.f;
  for (int hh = lane; hh < HSZ; hh += 32) {
    float p = tanhf(P1[(size_t)row * HSZ + hh] + q[(size_t)b * HSZ + hh]);
    sum += p * watt[hh];
  }
#pragma unroll
  for (int o = 16; o > 0; o >>= 1) sum += __shfl_xor(sum, o, 32);
  if (lane == 0) e[row] = sum + batt[0];
}

// ---------------- row softmax (in place), rows of length A, grid = B -----------
__global__ void k_softmax(float* __restrict__ e, int A) {
  int b = blockIdx.x, tid = threadIdx.x;
  __shared__ float red[128];
  float m = -1e30f;
  for (int a = tid; a < A; a += 128) m = fmaxf(m, e[b * A + a]);
  red[tid] = m; __syncthreads();
  for (int s = 64; s > 0; s >>= 1) { if (tid < s) red[tid] = fmaxf(red[tid], red[tid + s]); __syncthreads(); }
  m = red[0]; __syncthreads();
  float sum = 0.f;
  for (int a = tid; a < A; a += 128) { float v = __expf(e[b * A + a] - m); e[b * A + a] = v; sum += v; }
  red[tid] = sum; __syncthreads();
  for (int s = 64; s > 0; s >>= 1) { if (tid < s) red[tid] += red[tid + s]; __syncthreads(); }
  float inv = 1.0f / red[0];
  for (int a = tid; a < A; a += 128) e[b * A + a] *= inv;
}

// ---------------- context: ctx[b,f] = sum_a alpha[b,a]*feats[b,a,f] ------------
__global__ void k_context(const float* __restrict__ alpha, const _Float16* __restrict__ feats,
                          _Float16* __restrict__ ctx, int A) {
  int b = blockIdx.x;
  int f = blockIdx.y * blockDim.x + threadIdx.x;
  float sum = 0.f;
  for (int a = 0; a < A; ++a)
    sum += alpha[b * A + a] * (float)feats[((size_t)(b * A + a)) * HSZ + f];
  ctx[(size_t)b * HSZ + f] = (_Float16)sum;
}

// ---------------- log_softmax in place over V, grid = B*TDEC -------------------
__global__ void k_logsoftmax(float* __restrict__ x, int V) {
  float* p = x + (size_t)blockIdx.x * V;
  __shared__ float red[256];
  int tid = threadIdx.x;
  float m = -1e30f;
  for (int i = tid; i < V; i += 256) m = fmaxf(m, p[i]);
  red[tid] = m; __syncthreads();
  for (int s = 128; s > 0; s >>= 1) { if (tid < s) red[tid] = fmaxf(red[tid], red[tid + s]); __syncthreads(); }
  m = red[0]; __syncthreads();
  float sum = 0.f;
  for (int i = tid; i < V; i += 256) sum += __expf(p[i] - m);
  red[tid] = sum; __syncthreads();
  for (int s = 128; s > 0; s >>= 1) { if (tid < s) red[tid] += red[tid + s]; __syncthreads(); }
  float lse = m + __logf(red[0]);
  for (int i = tid; i < V; i += 256) p[i] -= lse;
}

// =================================================================================
extern "C" void kernel_launch(void* const* d_in, const int* in_sizes, int n_in,
                              void* d_out, int out_size, void* d_ws, size_t ws_size,
                              hipStream_t stream) {
  (void)in_sizes; (void)n_in; (void)out_size; (void)ws_size;
  const int*   code      = (const int*)d_in[0];
  const int*   comment   = (const int*)d_in[1];
  const float* embed     = (const float*)d_in[3];
  const float* enc_Wi    = (const float*)d_in[4];
  const float* enc_bi    = (const float*)d_in[5];
  const float* enc_Wh    = (const float*)d_in[6];
  const float* enc_bh    = (const float*)d_in[7];
  const float* rev_Wh    = (const float*)d_in[8];
  const float* rev_bh    = (const float*)d_in[9];
  const float* rev_Wa    = (const float*)d_in[10];
  const float* rev_ba    = (const float*)d_in[11];
  const float* rev_Wa2a  = (const float*)d_in[12];
  const float* rev_ba2a  = (const float*)d_in[13];
  const float* rev_Wh2a  = (const float*)d_in[14];
  const float* rev_bh2a  = (const float*)d_in[15];
  const float* rev_Watt  = (const float*)d_in[16];
  const float* rev_batt  = (const float*)d_in[17];
  const float* dec_Wi    = (const float*)d_in[18];
  const float* dec_bi    = (const float*)d_in[19];
  const float* dec_Wh    = (const float*)d_in[20];
  const float* dec_bh    = (const float*)d_in[21];
  const float* dec_Wa    = (const float*)d_in[22];
  const float* dec_ba    = (const float*)d_in[23];
  const float* dec_Wa2a  = (const float*)d_in[24];
  const float* dec_ba2a  = (const float*)d_in[25];
  const float* dec_Wh2a  = (const float*)d_in[26];
  const float* dec_bh2a  = (const float*)d_in[27];
  const float* dec_Watt  = (const float*)d_in[28];
  const float* dec_batt  = (const float*)d_in[29];
  const float* logit_W   = (const float*)d_in[30];
  const float* logit_b   = (const float*)d_in[31];
  float* out = (float*)d_out;

  // -------- workspace bump allocator --------
  char* ws = (char*)d_ws;
  size_t off = 0;
  auto alloc = [&](size_t bytes) -> void* {
    void* p = ws + off;
    off = (off + bytes + 255) & ~(size_t)255;
    return p;
  };
  const size_t WBIG  = (size_t)LSZ * L5;           // 512*2560
  const size_t WSQ   = (size_t)LSZ * HSZ;          // 512*512
  const int    VPAD  = ((VOUT + 255) / 256) * 256; // 30208
  const size_t WLOG  = (size_t)LSZ * VPAD;

  _Float16* w_encWi   = (_Float16*)alloc(WBIG * 2);
  _Float16* w_encWh   = (_Float16*)alloc(WBIG * 2);
  _Float16* w_revWh   = (_Float16*)alloc(RSZ * WBIG * 2);
  _Float16* w_revWa   = (_Float16*)alloc(RSZ * WBIG * 2);
  _Float16* w_revWa2a = (_Float16*)alloc(RSZ * WSQ * 2);
  _Float16* w_revWh2a = (_Float16*)alloc(RSZ * WSQ * 2);
  _Float16* w_decWi   = (_Float16*)alloc(WBIG * 2);
  _Float16* w_decWh   = (_Float16*)alloc(WBIG * 2);
  _Float16* w_decWa   = (_Float16*)alloc(WBIG * 2);
  _Float16* w_decWa2a = (_Float16*)alloc(WSQ * 2);
  _Float16* w_decWh2a = (_Float16*)alloc(WSQ * 2);
  _Float16* w_logitW  = (_Float16*)alloc(WLOG * 2);

  _Float16* enc_x   = (_Float16*)alloc((size_t)TENC * BSZ * ESZ * 2);  // rows [t*B+b]
  _Float16* dec_x   = (_Float16*)alloc((size_t)TDEC * BSZ * ESZ * 2);
  float*    XWi_enc = (float*)alloc((size_t)TENC * BSZ * L5 * 4);
  float*    XWi_dec = (float*)alloc((size_t)TDEC * BSZ * L5 * 4);
  _Float16* enc_hs  = (_Float16*)alloc((size_t)BSZ * TENC * LSZ * 2);  // [b][t][l]
  float*    P1      = (float*)alloc((size_t)BSZ * TENC * HSZ * 4);
  _Float16* thought = (_Float16*)alloc((size_t)BSZ * RSZ * LSZ * 2);   // [b][r][l]
  float*    TP      = (float*)alloc((size_t)BSZ * RSZ * HSZ * 4);
  _Float16* Hdec    = (_Float16*)alloc((size_t)BSZ * TDEC * LSZ * 2);  // [b][t][l]
  float*    sbuf    = (float*)alloc((size_t)BSZ * L5 * 4);
  _Float16* hbuf    = (_Float16*)alloc((size_t)BSZ * LSZ * 2);
  float*    cbuf    = (float*)alloc((size_t)BSZ * LSZ * 4);
  _Float16* h_enc   = (_Float16*)alloc((size_t)BSZ * LSZ * 2);
  float*    c_enc   = (float*)alloc((size_t)BSZ * LSZ * 4);
  float*    qbuf    = (float*)alloc((size_t)BSZ * HSZ * 4);
  _Float16* ctx     = (_Float16*)alloc((size_t)BSZ * LSZ * 2);
  float*    ebuf    = (float*)alloc((size_t)BSZ * TENC * 4);

  auto pack = [&](const float* src, _Float16* dst, int N, int K, int nsl) {
    int Npad = ((N + 255) / 256) * 256;
    size_t tot = (size_t)Npad * K * nsl;
    k_pack_w<<<dim3((unsigned)((tot + 255) / 256)), 256, 0, stream>>>(src, dst, N, Npad, K, nsl);
  };
  auto gemm = [&](const _Float16* A, int lda, const _Float16* Wp, int N, int K,
                  const float* bias, const float* aux, float* C, int ldc,
                  int M, int mode) {
    dim3 g((N + 255) / 256, M / 16);
    if (mode == 1)
      k_gemm_f16<1><<<g, 128, 0, stream>>>(A, lda, Wp, N, K, bias, aux, C, ldc);
    else if (mode == 2)
      k_gemm_f16<2><<<g, 128, 0, stream>>>(A, lda, Wp, N, K, bias, aux, C, ldc);
    else
      k_gemm_f16<0><<<g, 128, 0, stream>>>(A, lda, Wp, N, K, bias, aux, C, ldc);
  };

  // -------- weight repack to WMMA fragment order (parallel, once per launch) ---
  pack(enc_Wi,   w_encWi,   L5,   LSZ, 1);
  pack(enc_Wh,   w_encWh,   L5,   LSZ, 1);
  pack(rev_Wh,   w_revWh,   L5,   LSZ, RSZ);
  pack(rev_Wa,   w_revWa,   L5,   LSZ, RSZ);
  pack(rev_Wa2a, w_revWa2a, HSZ,  LSZ, RSZ);
  pack(rev_Wh2a, w_revWh2a, HSZ,  LSZ, RSZ);
  pack(dec_Wi,   w_decWi,   L5,   LSZ, 1);
  pack(dec_Wh,   w_decWh,   L5,   LSZ, 1);
  pack(dec_Wa,   w_decWa,   L5,   LSZ, 1);
  pack(dec_Wa2a, w_decWa2a, HSZ,  LSZ, 1);
  pack(dec_Wh2a, w_decWh2a, HSZ,  LSZ, 1);
  pack(logit_W,  w_logitW,  VOUT, LSZ, 1);

  // -------- embeddings (f16) --------
  k_embed<<<dim3(BSZ * TENC, ESZ / 256), 256, 0, stream>>>(code, embed, enc_x, TENC);
  k_embed<<<dim3(BSZ * TDEC, ESZ / 256), 256, 0, stream>>>(comment, embed, dec_x, TDEC);

  // -------- hoisted input-side GEMMs (big, parallel) --------
  gemm(enc_x, ESZ, w_encWi, L5, ESZ, enc_bi, nullptr, XWi_enc, L5, TENC * BSZ, 0);
  gemm(dec_x, ESZ, w_decWi, L5, ESZ, dec_bi, nullptr, XWi_dec, L5, TDEC * BSZ, 0);

  // -------- encoder recurrence --------
  hipMemsetAsync(hbuf, 0, (size_t)BSZ * LSZ * 2, stream);
  hipMemsetAsync(cbuf, 0, (size_t)BSZ * LSZ * 4, stream);
  for (int t = 0; t < TENC; ++t) {
    gemm(hbuf, LSZ, w_encWh, L5, LSZ, enc_bh, XWi_enc + (size_t)t * BSZ * L5,
         sbuf, L5, BSZ, 1);
    k_lstm<<<(BSZ * LSZ) / 256, 256, 0, stream>>>(sbuf, cbuf, hbuf,
                                                  enc_hs + (size_t)t * LSZ, TENC * LSZ);
  }
  hipMemcpyAsync(h_enc, hbuf, (size_t)BSZ * LSZ * 2, hipMemcpyDeviceToDevice, stream);
  hipMemcpyAsync(c_enc, cbuf, (size_t)BSZ * LSZ * 4, hipMemcpyDeviceToDevice, stream);

  // -------- reviewer (8 steps, per-step params) --------
  for (int r = 0; r < RSZ; ++r) {
    gemm(enc_hs, LSZ, w_revWa2a + (size_t)r * WSQ, HSZ, LSZ,
         rev_ba2a + r * HSZ, nullptr, P1, HSZ, BSZ * TENC, 0);
    gemm(hbuf, LSZ, w_revWh2a + (size_t)r * WSQ, HSZ, LSZ,
         rev_bh2a + r * HSZ, nullptr, qbuf, HSZ, BSZ, 0);
    k_att_scores<<<BSZ * TENC, 32, 0, stream>>>(P1, qbuf, rev_Watt + r * HSZ,
                                                rev_batt + r, ebuf, TENC);
    k_softmax<<<BSZ, 128, 0, stream>>>(ebuf, TENC);
    k_context<<<dim3(BSZ, HSZ / 256), 256, 0, stream>>>(ebuf, enc_hs, ctx, TENC);
    gemm(hbuf, LSZ, w_revWh + (size_t)r * WBIG, L5, LSZ, rev_bh + r * L5,
         nullptr, sbuf, L5, BSZ, 0);
    gemm(ctx, LSZ, w_revWa + (size_t)r * WBIG, L5, LSZ, rev_ba + r * L5,
         nullptr, sbuf, L5, BSZ, 2);
    k_lstm<<<(BSZ * LSZ) / 256, 256, 0, stream>>>(sbuf, cbuf, hbuf,
                                                  thought + (size_t)r * LSZ, RSZ * LSZ);
  }

  // -------- decoder starts from encoder final state --------
  hipMemcpyAsync(hbuf, h_enc, (size_t)BSZ * LSZ * 2, hipMemcpyDeviceToDevice, stream);
  hipMemcpyAsync(cbuf, c_enc, (size_t)BSZ * LSZ * 4, hipMemcpyDeviceToDevice, stream);

  // thought-side attention projection (hoisted)
  gemm(thought, LSZ, w_decWa2a, HSZ, LSZ, dec_ba2a, nullptr, TP, HSZ, BSZ * RSZ, 0);

  for (int t = 0; t < TDEC; ++t) {
    gemm(hbuf, LSZ, w_decWh2a, HSZ, LSZ, dec_bh2a, nullptr, qbuf, HSZ, BSZ, 0);
    k_att_scores<<<BSZ * RSZ, 32, 0, stream>>>(TP, qbuf, dec_Watt, dec_batt, ebuf, RSZ);
    k_softmax<<<BSZ, 128, 0, stream>>>(ebuf, RSZ);
    k_context<<<dim3(BSZ, HSZ / 256), 256, 0, stream>>>(ebuf, thought, ctx, RSZ);
    gemm(hbuf, LSZ, w_decWh, L5, LSZ, dec_bh, XWi_dec + (size_t)t * BSZ * L5,
         sbuf, L5, BSZ, 1);
    gemm(ctx, LSZ, w_decWa, L5, LSZ, dec_ba, nullptr, sbuf, L5, BSZ, 2);
    k_lstm<<<(BSZ * LSZ) / 256, 256, 0, stream>>>(sbuf, cbuf, hbuf,
                                                  Hdec + (size_t)t * LSZ, TDEC * LSZ);
  }

  // -------- big logit GEMM straight into d_out, then in-place log_softmax ------
  gemm(Hdec, LSZ, w_logitW, VOUT, LSZ, logit_b, nullptr, out, VOUT, BSZ * TDEC, 0);
  k_logsoftmax<<<BSZ * TDEC, 256, 0, stream>>>(out, VOUT);
}
